// MultiHeadAttention_47923245089231
// MI455X (gfx1250) — matrix operand order repaired and checked by reading, not compiled, not run on hardware
//
#include <hip/hip_runtime.h>
#include <stdint.h>

// MHA: B=8,H=8,S=1024,D=64. fp32 in/out, bf16 WMMA internally.
// ws: q_bf16 | k_bf16 | v_bf16, each 65536*64 u16 (25.2 MB total).

typedef __attribute__((ext_vector_type(16))) __bf16 bf16x16;
typedef __attribute__((ext_vector_type(8)))  float   f32x8;
typedef int v4i __attribute__((vector_size(16)));
typedef v4i __attribute__((address_space(1)))* gv4i_p;   // global int4*
typedef v4i __attribute__((address_space(3)))* lv4i_p;   // LDS int4*

union Frag16 { uint16_t h[16]; __bf16 b[16]; uint4 q[2]; bf16x16 v; };

#if __has_builtin(__builtin_amdgcn_global_load_async_to_lds_b128)
#define HAVE_ASYNC_LDS 1
#else
#define HAVE_ASYNC_LDS 0
#endif

__device__ __forceinline__ uint16_t f2bf(float f) {   // native v_cvt path (RNE)
  return __builtin_bit_cast(uint16_t, (__bf16)f);
}
__device__ __forceinline__ f32x8 zero8() {
  f32x8 z;
#pragma unroll
  for (int i = 0; i < 8; ++i) z[i] = 0.f;
  return z;
}
__device__ __forceinline__ f32x8 wmma_bf16(bf16x16 a, bf16x16 b, f32x8 c) {
  return __builtin_amdgcn_wmma_f32_16x16x32_bf16(false, a, false, b, (short)0, c,
                                                 false, false);
}

// ---------------- Kernel 1: y = relu(x @ W^T + b) -> bf16 workspace ----------
__global__ __launch_bounds__(256) void proj_kernel(
    const float* __restrict__ xq, const float* __restrict__ xk,
    const float* __restrict__ xv,
    const float* __restrict__ Wq, const float* __restrict__ Bq,
    const float* __restrict__ Wk, const float* __restrict__ Bk,
    const float* __restrict__ Wv, const float* __restrict__ Bv,
    uint16_t* __restrict__ ws) {
  __shared__ __align__(16) uint16_t w_lds[64 * 64];
  __shared__ float bias_lds[64];

  const int tilesPerTensor = 65536 / 128;          // 512
  int tid  = blockIdx.x / tilesPerTensor;          // 0=q 1=k 2=v
  int tile = blockIdx.x % tilesPerTensor;
  const float *x, *W, *bias; uint16_t* out;
  if (tid == 0)      { x = xq; W = Wq; bias = Bq; out = ws; }
  else if (tid == 1) { x = xk; W = Wk; bias = Bk; out = ws + (size_t)65536 * 64; }
  else               { x = xv; W = Wv; bias = Bv; out = ws + (size_t)2 * 65536 * 64; }

  int t = threadIdx.x;
  { // W (64x64 f32) -> LDS bf16, 16 elems/thread
    int base = t * 16;
    const float4* src = (const float4*)(W + base);
#pragma unroll
    for (int g = 0; g < 4; ++g) {
      float4 f = src[g];
      w_lds[base + g * 4 + 0] = f2bf(f.x);
      w_lds[base + g * 4 + 1] = f2bf(f.y);
      w_lds[base + g * 4 + 2] = f2bf(f.z);
      w_lds[base + g * 4 + 3] = f2bf(f.w);
    }
  }
  if (t < 64) bias_lds[t] = bias[t];
  __syncthreads();

  int wave = t >> 5, lane = t & 31, hi = (lane >> 4) & 1, ln = lane & 15;
  int rowBase = tile * 128 + wave * 16;

  // A fragments for this wave's 16 rows, K = 0..63 in two steps
  Frag16 afr[2];
  const float* xrow = x + (size_t)(rowBase + ln) * 64;
#pragma unroll
  for (int kk = 0; kk < 2; ++kk) {
    int d0 = kk * 32 + hi * 16;
#pragma unroll
    for (int g = 0; g < 2; ++g) {                  // elems [0..7] at d0, [8..15] at d0+8
      const float4* s = (const float4*)(xrow + d0 + g * 8);
      float4 f0 = s[0], f1 = s[1];
      int e = g * 8;
      afr[kk].b[e + 0] = (__bf16)f0.x; afr[kk].b[e + 1] = (__bf16)f0.y;
      afr[kk].b[e + 2] = (__bf16)f0.z; afr[kk].b[e + 3] = (__bf16)f0.w;
      afr[kk].b[e + 4] = (__bf16)f1.x; afr[kk].b[e + 5] = (__bf16)f1.y;
      afr[kk].b[e + 6] = (__bf16)f1.z; afr[kk].b[e + 7] = (__bf16)f1.w;
    }
  }

#pragma unroll
  for (int nt = 0; nt < 4; ++nt) {
    f32x8 c = zero8();
#pragma unroll
    for (int kk = 0; kk < 2; ++kk) {
      Frag16 b;
      const uint4* s = (const uint4*)(w_lds + (nt * 16 + ln) * 64 + kk * 32 + hi * 16);
      b.q[0] = s[0]; b.q[1] = s[1];
      c = wmma_bf16(afr[kk].v, b.v, c);
    }
    int n = nt * 16 + ln;
    float bv = bias_lds[n];
#pragma unroll
    for (int r = 0; r < 8; ++r) {
      float v = c[r] + bv;
      v = v > 0.f ? v : 0.f;
      out[(size_t)(rowBase + r + 8 * hi) * 64 + n] = f2bf(v);
    }
  }
}

// ---------------- Kernel 2: flash attention + permute + residual -------------
__global__ __launch_bounds__(256) void attn_kernel(
    const uint16_t* __restrict__ ws, const float* __restrict__ queries,
    float* __restrict__ outp) {
  __shared__ __align__(16) uint16_t k_lds[64 * 64];    // [j][d] row-major
  __shared__ __align__(16) uint16_t vT_lds[64 * 64];   // [d][j] (transposed)
  __shared__ __align__(16) uint16_t p_lds[8 * 16 * 64];

  const uint16_t* q_ws = ws;
  const uint16_t* k_ws = ws + (size_t)65536 * 64;
  const uint16_t* v_ws = ws + (size_t)2 * 65536 * 64;

  int bh = blockIdx.x >> 3;
  int q0 = (blockIdx.x & 7) * 128;
  int t = threadIdx.x, wave = t >> 5, lane = t & 31, hi = (lane >> 4) & 1,
      ln = lane & 15;
  int qrow0 = q0 + wave * 16;

  // Q fragments (persistent)
  Frag16 qfr[2];
  const uint16_t* qrow = q_ws + ((size_t)bh * 1024 + qrow0 + ln) * 64;
#pragma unroll
  for (int kk = 0; kk < 2; ++kk) {
    qfr[kk].q[0] = *(const uint4*)(qrow + kk * 32 + hi * 16);
    qfr[kk].q[1] = *(const uint4*)(qrow + kk * 32 + hi * 16 + 8);
  }

  float m_i[8], l_i[8];
  f32x8 acc[4];
#pragma unroll
  for (int r = 0; r < 8; ++r) { m_i[r] = -1e30f; l_i[r] = 0.f; }
#pragma unroll
  for (int dt = 0; dt < 4; ++dt) acc[dt] = zero8();

  uint16_t* pbuf = p_lds + wave * 1024;
  int j = t >> 2, part = t & 3;                       // cooperative-load mapping
  const uint16_t* kbase = k_ws + ((size_t)bh * 1024 + j) * 64 + part * 16;
  const uint16_t* vbase = v_ws + ((size_t)bh * 1024 + j) * 64 + part * 16;
  uint16_t* kdst16 = k_lds + j * 64 + part * 16;

  for (int kt = 0; kt < 16; ++kt) {
    __syncthreads();
    { // K tile: async global->LDS if available; V tile: regs + transposed stores
      const uint16_t* ksrc = kbase + (size_t)kt * 64 * 64;
#if HAVE_ASYNC_LDS
      __builtin_amdgcn_global_load_async_to_lds_b128(
          (gv4i_p)ksrc, (lv4i_p)kdst16, 0, 0);
      __builtin_amdgcn_global_load_async_to_lds_b128(
          (gv4i_p)(ksrc + 8), (lv4i_p)(kdst16 + 8), 0, 0);
#else
      const uint4* ks4 = (const uint4*)ksrc;
      uint4* kd4 = (uint4*)kdst16;
      kd4[0] = ks4[0]; kd4[1] = ks4[1];
#endif
      const uint4* vsrc = (const uint4*)(vbase + (size_t)kt * 64 * 64);
      Frag16 vv; vv.q[0] = vsrc[0]; vv.q[1] = vsrc[1];
#pragma unroll
      for (int i = 0; i < 16; ++i) vT_lds[(part * 16 + i) * 64 + j] = vv.h[i];
      if (kt < 15) {  // prefetch next tile -> global_prefetch_b8
        __builtin_prefetch(kbase + (size_t)(kt + 1) * 64 * 64, 0, 1);
        __builtin_prefetch(vbase + (size_t)(kt + 1) * 64 * 64, 0, 1);
      }
#if HAVE_ASYNC_LDS
      asm volatile("s_wait_asynccnt 0" ::: "memory");
#endif
    }
    __syncthreads();

    // S = Q K^T
    f32x8 s[4];
#pragma unroll
    for (int nt = 0; nt < 4; ++nt) {
      f32x8 c = zero8();
#pragma unroll
      for (int kk = 0; kk < 2; ++kk) {
        Frag16 b;
        const uint4* src =
            (const uint4*)(k_lds + (nt * 16 + ln) * 64 + kk * 32 + hi * 16);
        b.q[0] = src[0]; b.q[1] = src[1];
        c = wmma_bf16(qfr[kk].v, b.v, c);
      }
      s[nt] = c;
    }

    // online softmax (scale = 1/sqrt(64))
#pragma unroll
    for (int r = 0; r < 8; ++r) {
      float mx = -1e30f;
#pragma unroll
      for (int nt = 0; nt < 4; ++nt) {
        s[nt][r] *= 0.125f;
        mx = fmaxf(mx, s[nt][r]);
      }
#pragma unroll
      for (int m = 1; m < 16; m <<= 1) mx = fmaxf(mx, __shfl_xor(mx, m, 32));
      float mnew  = fmaxf(m_i[r], mx);
      float alpha = __expf(m_i[r] - mnew);
      float rs = 0.f;
#pragma unroll
      for (int nt = 0; nt < 4; ++nt) {
        float p = __expf(s[nt][r] - mnew);
        s[nt][r] = p;
        rs += p;
      }
#pragma unroll
      for (int m = 1; m < 16; m <<= 1) rs += __shfl_xor(rs, m, 32);
      l_i[r] = l_i[r] * alpha + rs;
      m_i[r] = mnew;
#pragma unroll
      for (int dt = 0; dt < 4; ++dt) acc[dt][r] *= alpha;
    }

    // P -> LDS (C/D layout -> row-major) for re-load in A layout
#pragma unroll
    for (int nt = 0; nt < 4; ++nt)
#pragma unroll
      for (int r = 0; r < 8; ++r)
        pbuf[(r + 8 * hi) * 64 + nt * 16 + ln] = f2bf(s[nt][r]);
    __syncthreads();

    // O += P V
    Frag16 afr[2];
#pragma unroll
    for (int kk = 0; kk < 2; ++kk) {
      afr[kk].q[0] = *(const uint4*)(pbuf + ln * 64 + kk * 32 + hi * 16);
      afr[kk].q[1] = *(const uint4*)(pbuf + ln * 64 + kk * 32 + hi * 16 + 8);
    }
#pragma unroll
    for (int dt = 0; dt < 4; ++dt) {
#pragma unroll
      for (int kk = 0; kk < 2; ++kk) {
        Frag16 b;
        const uint4* src =
            (const uint4*)(vT_lds + (dt * 16 + ln) * 64 + kk * 32 + hi * 16);
        b.q[0] = src[0]; b.q[1] = src[1];
        acc[dt] = wmma_bf16(afr[kk].v, b.v, acc[dt]);
      }
    }
  }

  // epilogue: normalize, apply (0,1,3,2)-transpose-reshape, add residual
  float inv[8];
#pragma unroll
  for (int r = 0; r < 8; ++r) inv[r] = 1.f / l_i[r];
  int sa_hi = qrow0 >> 6;    // (s_a >> 6), constant per wave
  int sa_lo = qrow0 & 63;
  const float* qin = queries + (size_t)bh * 1024 * 64;
  float* outb = outp + (size_t)bh * 1024 * 64;
#pragma unroll
  for (int dt = 0; dt < 4; ++dt) {
    int da = dt * 16 + ln;
    int so = da * 16 + sa_hi;
#pragma unroll
    for (int r = 0; r < 8; ++r) {
      int dob = sa_lo + r + 8 * hi;
      size_t idx = (size_t)so * 64 + dob;
      outb[idx] = acc[dt][r] * inv[r] + qin[idx];
    }
  }
}

extern "C" void kernel_launch(void* const* d_in, const int* in_sizes, int n_in,
                              void* d_out, int out_size, void* d_ws,
                              size_t ws_size, hipStream_t stream) {
  (void)in_sizes; (void)n_in; (void)out_size; (void)ws_size;
  const float* q  = (const float*)d_in[0];
  const float* k  = (const float*)d_in[1];
  const float* v  = (const float*)d_in[2];
  const float* Wq = (const float*)d_in[3];
  const float* Bq = (const float*)d_in[4];
  const float* Wk = (const float*)d_in[5];
  const float* Bk = (const float*)d_in[6];
  const float* Wv = (const float*)d_in[7];
  const float* Bv = (const float*)d_in[8];
  uint16_t* ws = (uint16_t*)d_ws;
  float* out = (float*)d_out;

  proj_kernel<<<dim3(3 * 512), dim3(256), 0, stream>>>(q, k, v, Wq, Bq, Wk, Bk,
                                                       Wv, Bv, ws);
  attn_kernel<<<dim3(512), dim3(256), 0, stream>>>(ws, q, out);
}
